// ResidualRoPETransformer_27693949125142
// MI455X (gfx1250) — matrix operand order repaired and checked by reading, not compiled, not run on hardware
//
#include <hip/hip_runtime.h>
#include <hip/hip_bf16.h>
#include <math.h>

typedef _Float16 v16h __attribute__((ext_vector_type(16)));
typedef _Float16 v8h  __attribute__((ext_vector_type(8)));
typedef float    v8f  __attribute__((ext_vector_type(8)));

#define EMB 256
#define HID 256
#define NH 8
#define HD 32
#define SEQ 512
#define BU 64
#define MTOT (BU * SEQ)          // 32768 tokens
#define QKV_N (3 * HID)          // 768

// ---------------------------------------------------------------------------
// Fragment loaders (wave32 WMMA 16x16x32 f16; contiguous k order k = 16*(lane/16) + s, not the ISA table order)
// ---------------------------------------------------------------------------

// A operand: 16x32 (MxK). lane<16: row=lane, K 0..15;
// lane>=16: row=lane-16, K 16..31. One 32B load.
__device__ __forceinline__ v16h load_fragA(const _Float16* __restrict__ base, int ld) {
    const int lane = threadIdx.x & 31;
    const int row  = lane & 15;
    const int kh   = (lane >> 4) << 4;          // 0 or 16
    const _Float16* p = base + (size_t)row * ld + kh;
    return *(const v16h*)(p);
}

// B operand: 32x16 (KxN). lane holds column lane&15; K contiguous:
// lanes 0-15 -> K 0..15, lanes 16-31 -> K 16..31. One 32B load.
// base points to row-major [N][K] storage of B^T (i.e. W[N,K]).
__device__ __forceinline__ v16h load_fragB(const _Float16* __restrict__ base, int ld) {
    const int lane = threadIdx.x & 31;
    const int col  = lane & 15;
    const int ko   = (lane >> 4) << 4;          // 0 or 16
    return *(const v16h*)(base + (size_t)col * ld + ko);
}

__device__ __forceinline__ v8f wmma_f16(v16h a, v16h b, v8f c) {
    return __builtin_amdgcn_wmma_f32_16x16x32_f16(false, a, false, b, (short)0, c, false, false);
}

// 16-lane xor-butterfly max. v_permlane16_b32 keeps it on the VALU pipe
// (one instruction per step) instead of ds_bpermute on the LDS pipe.
#if __has_builtin(__builtin_amdgcn_permlane16)
__device__ __forceinline__ float pl16_max(float x, int slo, int shi) {
    int xi = __builtin_bit_cast(int, x);
    int yi = __builtin_amdgcn_permlane16(xi, xi, slo, shi, false, false);
    return fmaxf(x, __builtin_bit_cast(float, yi));
}
__device__ __forceinline__ float rowmax16(float x) {
    x = pl16_max(x, 0x67452301, 0xEFCDAB89);   // xor 1
    x = pl16_max(x, 0x45670123, 0xCDEF89AB);   // xor 3
    x = pl16_max(x, 0x01234567, 0x89ABCDEF);   // xor 7
    x = pl16_max(x, 0xFEDCBA98, 0x76543210);   // xor 8
    return x;
}
#else
__device__ __forceinline__ float rowmax16(float x) {
    for (int off = 1; off < 16; off <<= 1)
        x = fmaxf(x, __shfl_xor(x, off, 32));
    return x;
}
#endif

// ---------------------------------------------------------------------------
// Weight norm: out[n,k] = f16( g[n] * v[n,k] / ||v[n,:]|| ), K = 256
// ---------------------------------------------------------------------------
__global__ void wn_kernel(const float* __restrict__ v, const float* __restrict__ g,
                          _Float16* __restrict__ out) {
    __shared__ float red[256];
    const int row = blockIdx.x, t = threadIdx.x;
    float x = v[(size_t)row * 256 + t];
    red[t] = x * x;
    __syncthreads();
    for (int s = 128; s > 0; s >>= 1) {
        if (t < s) red[t] += red[t + s];
        __syncthreads();
    }
    float scale = g[row] * rsqrtf(red[0]);
    out[(size_t)row * 256 + t] = (_Float16)(x * scale);
}

// ---------------------------------------------------------------------------
// LayerNorm over E=256, f32 in -> f16 out
// ---------------------------------------------------------------------------
__global__ void ln_kernel(const float* __restrict__ x, const float* __restrict__ gamma,
                          const float* __restrict__ beta, _Float16* __restrict__ out) {
    __shared__ float red[256];
    const int row = blockIdx.x, t = threadIdx.x;
    float v = x[(size_t)row * 256 + t];
    red[t] = v;
    __syncthreads();
    for (int s = 128; s > 0; s >>= 1) { if (t < s) red[t] += red[t + s]; __syncthreads(); }
    float mu = red[0] * (1.0f / 256.0f);
    __syncthreads();
    float d = v - mu;
    red[t] = d * d;
    __syncthreads();
    for (int s = 128; s > 0; s >>= 1) { if (t < s) red[t] += red[t + s]; __syncthreads(); }
    float inv = rsqrtf(red[0] * (1.0f / 256.0f) + 1e-5f);
    out[(size_t)row * 256 + t] = (_Float16)(d * inv * gamma[t] + beta[t]);
}

// ---------------------------------------------------------------------------
// WMMA GEMM: C[M,N] = A[M,K](f16) @ W[N,K]^T(f16) + bias.
// Block = 256 threads (8 waves). Each wave computes a 16x32 tile
// (two 16x16 WMMA accumulators sharing one A fragment -> half the A loads).
//   MODE 0: f16 store   MODE 1: exact-GELU -> f16   MODE 2: f32 (+residual)
// ---------------------------------------------------------------------------
template <int MODE>
__global__ void gemm_kernel(const _Float16* __restrict__ A, const _Float16* __restrict__ W,
                            const float* __restrict__ bias, const float* __restrict__ residual,
                            _Float16* __restrict__ outh, float* __restrict__ outf,
                            int M, int N, int K) {
    const int wave  = threadIdx.x >> 5;
    const int lane  = threadIdx.x & 31;
    const int mtile = blockIdx.x;
    const int ntile = blockIdx.y * 16 + wave * 2;        // two N-tiles per wave
    if (ntile * 16 >= N) return;

    const _Float16* Abase = A + (size_t)mtile * 16 * K;
    const _Float16* W0 = W + (size_t)ntile * 16 * K;
    const _Float16* W1 = W0 + (size_t)16 * K;

    v8f acc0 = {}, acc1 = {};
#pragma unroll
    for (int kc = 0; kc < 8; ++kc) {           // K = 256 = 8 x 32
        v16h a  = load_fragA(Abase + kc * 32, K);
        v16h b0 = load_fragB(W0 + kc * 32, K);
        v16h b1 = load_fragB(W1 + kc * 32, K);
        acc0 = wmma_f16(a, b0, acc0);
        acc1 = wmma_f16(a, b1, acc1);
    }

    const int n0 = ntile * 16 + (lane & 15);
    const float bn0 = bias[n0];
    const float bn1 = bias[n0 + 16];
#pragma unroll
    for (int v = 0; v < 8; ++v) {
        const int m = mtile * 16 + v + ((lane >> 4) << 3);
        float x0 = acc0[v] + bn0;
        float x1 = acc1[v] + bn1;
        if (MODE == 0) {
            outh[(size_t)m * N + n0]      = (_Float16)x0;
            outh[(size_t)m * N + n0 + 16] = (_Float16)x1;
        } else if (MODE == 1) {
            float g0 = 0.5f * x0 * (1.0f + erff(x0 * 0.70710678118654752f));
            float g1 = 0.5f * x1 * (1.0f + erff(x1 * 0.70710678118654752f));
            outh[(size_t)m * N + n0]      = (_Float16)g0;
            outh[(size_t)m * N + n0 + 16] = (_Float16)g1;
        } else {
            outf[(size_t)m * N + n0]      = x0 + residual[(size_t)m * N + n0];
            outf[(size_t)m * N + n0 + 16] = x1 + residual[(size_t)m * N + n0 + 16];
        }
    }
}

// ---------------------------------------------------------------------------
// RoPE (in place on qkv f16) + scatter V into vT[head][dim][pos] (f16)
// qkv layout: [token][h*96 + (q:0..31 | k:32..63 | v:64..95)]
// ---------------------------------------------------------------------------
__global__ void rope_kernel(_Float16* __restrict__ qkv, _Float16* __restrict__ vT) {
    const int idx = blockIdx.x * blockDim.x + threadIdx.x;   // MTOT*NH*16
    const int i = idx & 15;               // rotary pair
    const int h = (idx >> 4) & 7;
    const int t = idx >> 7;
    if (t >= MTOT) return;
    const int a  = t & (SEQ - 1);
    const int bu = t >> 9;

    _Float16* base = qkv + (size_t)t * QKV_N + h * 96;
    const float freq = expf(-(float)(2 * i) * (9.210340371976184f / 32.0f)); // 10000^(-2i/32)
    const float ang = (float)a * freq;
    const float c = cosf(ang), s = sinf(ang);

    float q0 = (float)base[2 * i], q1 = (float)base[2 * i + 1];
    base[2 * i]     = (_Float16)(q0 * c - q1 * s);
    base[2 * i + 1] = (_Float16)(q1 * c + q0 * s);
    float k0 = (float)base[32 + 2 * i], k1 = (float)base[32 + 2 * i + 1];
    base[32 + 2 * i]     = (_Float16)(k0 * c - k1 * s);
    base[32 + 2 * i + 1] = (_Float16)(k1 * c + k0 * s);

    // transpose V for P@V B-operand loads: vT[(bu*NH+h)*32 + d][a]
    const size_t hb = ((size_t)(bu * NH + h)) * HD * SEQ;
    vT[hb + (size_t)(2 * i)     * SEQ + a] = base[64 + 2 * i];
    vT[hb + (size_t)(2 * i + 1) * SEQ + a] = base[64 + 2 * i + 1];
}

// ---------------------------------------------------------------------------
// Flash attention. Grid (BU*NH, SEQ/16/8), block 256 = 8 waves.
// Software-pipelined: K/V fragments for chunk kc+1 are issued right after
// the score WMMAs of chunk kc, so the online-softmax VALU work (permlane
// max butterflies, exp) and the P LDS roundtrip overlap the global loads.
// Row sums are accumulated as P @ ones on the matrix pipe.
// ---------------------------------------------------------------------------
__global__ void attn_kernel(_Float16* __restrict__ qkv, const _Float16* __restrict__ vT,
                            _Float16* __restrict__ attn_out) {
    __shared__ _Float16 smem[8 * 16 * 32];
    const int wave = threadIdx.x >> 5;
    const int lane = threadIdx.x & 31;
    const int bu = blockIdx.x >> 3;
    const int h  = blockIdx.x & 7;
    const int qtile = blockIdx.y * 8 + wave;
    const int row0 = bu * SEQ + qtile * 16;        // first token of q tile
    _Float16* myP = &smem[wave * 16 * 32];

    const _Float16* qbase = qkv + (size_t)row0 * QKV_N + h * 96;
    v16h qf = load_fragA(qbase, QKV_N);            // 16x32 query tile

    const _Float16* khead = qkv + (size_t)(bu * SEQ) * QKV_N + h * 96 + 32;
    const _Float16* vhead = vT + ((size_t)(bu * NH + h)) * HD * SEQ;
    const float scale = 0.17677669529663688f;      // 1/sqrt(32)

    v16h onesf;
#pragma unroll
    for (int i = 0; i < 16; ++i) onesf[i] = (_Float16)1.0f;

    float m_i[8];
#pragma unroll
    for (int v = 0; v < 8; ++v) m_i[v] = -1e30f;
    v8f acc0 = {}, acc1 = {}, accl = {};           // accl = running row sums

    // prologue: fragments for chunk 0
    v16h kf0 = load_fragB(khead, QKV_N);
    v16h kf1 = load_fragB(khead + (size_t)16 * QKV_N, QKV_N);
    v16h vf0 = load_fragB(vhead, SEQ);
    v16h vf1 = load_fragB(vhead + (size_t)16 * SEQ, SEQ);

    for (int kc = 0; kc < 16; ++kc) {              // 32 keys / iter
        v8f z = {};
        v8f s0 = wmma_f16(qf, kf0, z);
        v8f s1 = wmma_f16(qf, kf1, z);

        // issue next chunk's loads now; consumed next iteration
        v16h nkf0 = kf0, nkf1 = kf1, nvf0 = vf0, nvf1 = vf1;
        if (kc < 15) {
            const int k1 = (kc + 1) * 32;
            const _Float16* nk = khead + (size_t)k1 * QKV_N;
            nkf0 = load_fragB(nk, QKV_N);
            nkf1 = load_fragB(nk + (size_t)16 * QKV_N, QKV_N);
            nvf0 = load_fragB(vhead + k1, SEQ);
            nvf1 = load_fragB(vhead + (size_t)16 * SEQ + k1, SEQ);
        }

        const int half = lane >> 4;
        const int col  = lane & 15;
#pragma unroll
        for (int v = 0; v < 8; ++v) {
            float a0 = s0[v] * scale, a1 = s1[v] * scale;
            float mn = fmaxf(m_i[v], rowmax16(fmaxf(a0, a1)));
            float p0 = __expf(a0 - mn), p1 = __expf(a1 - mn);
            float alpha = __expf(m_i[v] - mn);
            m_i[v] = mn;
            acc0[v] *= alpha;
            acc1[v] *= alpha;
            accl[v] *= alpha;
            const int prow = v + (half << 3);
            myP[prow * 32 + col]      = (_Float16)p0;
            myP[prow * 32 + 16 + col] = (_Float16)p1;
        }
        v16h pf = load_fragA(myP, 32);             // P 16x32 as A operand
        acc0 = wmma_f16(pf, vf0, acc0);
        acc1 = wmma_f16(pf, vf1, acc1);
        accl = wmma_f16(pf, onesf, accl);          // row sums on matrix pipe

        kf0 = nkf0; kf1 = nkf1; vf0 = nvf0; vf1 = nvf1;
    }

    const int half = lane >> 4;
    const int col  = lane & 15;
#pragma unroll
    for (int v = 0; v < 8; ++v) {
        float inv = 1.0f / accl[v];                // every column holds the row sum
        const int tok = row0 + v + (half << 3);
        attn_out[(size_t)tok * HID + h * HD + col]      = (_Float16)(acc0[v] * inv);
        attn_out[(size_t)tok * HID + h * HD + 16 + col] = (_Float16)(acc1[v] * inv);
    }
}

// ---------------------------------------------------------------------------
// Host-side orchestration
// ---------------------------------------------------------------------------
extern "C" void kernel_launch(void* const* d_in, const int* in_sizes, int n_in,
                              void* d_out, int out_size, void* d_ws, size_t ws_size,
                              hipStream_t stream) {
    const float* z      = (const float*)d_in[0];
    const float* ln1_g  = (const float*)d_in[1];
    const float* ln1_b  = (const float*)d_in[2];
    const float* proj_v = (const float*)d_in[3];
    const float* proj_g = (const float*)d_in[4];
    const float* proj_b = (const float*)d_in[5];
    const float* ff_v   = (const float*)d_in[6];
    const float* ff_g   = (const float*)d_in[7];
    const float* ff_b   = (const float*)d_in[8];
    const float* ln2_g  = (const float*)d_in[9];
    const float* ln2_b  = (const float*)d_in[10];
    const float* w1_v   = (const float*)d_in[11];
    const float* w1_g   = (const float*)d_in[12];
    const float* w1_b   = (const float*)d_in[13];
    const float* w2_v   = (const float*)d_in[14];
    const float* w2_g   = (const float*)d_in[15];
    const float* w2_b   = (const float*)d_in[16];
    float* out = (float*)d_out;

    char* ws = (char*)d_ws;
    _Float16* Wqkv = (_Float16*)(ws + 0);                      // 768*256 f16
    _Float16* Wo   = (_Float16*)(ws + 393216);                 // 256*256
    _Float16* W1   = (_Float16*)(ws + 524288);
    _Float16* W2   = (_Float16*)(ws + 655360);
    _Float16* qkv  = (_Float16*)(ws + 786432);                 // 32768*768 f16 (reused as h1)
    _Float16* zln  = (_Float16*)(ws + 786432 + 50331648ull);   // 32768*256 f16 (reused as hln)
    _Float16* aout = (_Float16*)(ws + 786432 + 50331648ull + 16777216ull);
    _Float16* vT   = (_Float16*)(ws + 786432 + 50331648ull + 2 * 16777216ull);
    float*    z2   = (float*)   (ws + 786432 + 50331648ull + 3 * 16777216ull);
    _Float16* h1   = qkv;
    _Float16* hln  = zln;

    // 1. weight-norm -> f16 weights
    wn_kernel<<<QKV_N, 256, 0, stream>>>(proj_v, proj_g, Wqkv);
    wn_kernel<<<EMB,   256, 0, stream>>>(ff_v, ff_g, Wo);
    wn_kernel<<<EMB,   256, 0, stream>>>(w1_v, w1_g, W1);
    wn_kernel<<<EMB,   256, 0, stream>>>(w2_v, w2_g, W2);

    // 2. LN1
    ln_kernel<<<MTOT, 256, 0, stream>>>(z, ln1_g, ln1_b, zln);

    // 3. QKV projection: [32768,256] x [768,256]^T  (N covered 256/block)
    gemm_kernel<0><<<dim3(MTOT / 16, QKV_N / 256), 256, 0, stream>>>(
        zln, Wqkv, proj_b, nullptr, qkv, nullptr, MTOT, QKV_N, EMB);

    // 4. RoPE + V transpose
    rope_kernel<<<(MTOT * NH * 16) / 256, 256, 0, stream>>>(qkv, vT);

    // 5. flash attention
    attn_kernel<<<dim3(BU * NH, SEQ / 16 / 8), 256, 0, stream>>>(qkv, vT, aout);

    // 6. out-proj + bias + residual -> z2 (f32)
    gemm_kernel<2><<<dim3(MTOT / 16, EMB / 256), 256, 0, stream>>>(
        aout, Wo, ff_b, z, nullptr, z2, MTOT, EMB, HID);

    // 7. LN2
    ln_kernel<<<MTOT, 256, 0, stream>>>(z2, ln2_g, ln2_b, hln);

    // 8. FFN1 + GELU
    gemm_kernel<1><<<dim3(MTOT / 16, EMB / 256), 256, 0, stream>>>(
        hln, W1, w1_b, nullptr, h1, nullptr, MTOT, EMB, EMB);

    // 9. FFN2 + bias + residual -> out (f32)
    gemm_kernel<2><<<dim3(MTOT / 16, EMB / 256), 256, 0, stream>>>(
        h1, W2, w2_b, z2, nullptr, out, MTOT, EMB, EMB);
}
